// CausalSelfAttention_2284922601493
// MI455X (gfx1250) — matrix operand order repaired and checked by reading, not compiled, not run on hardware
//
#include <hip/hip_runtime.h>
#include <hip/hip_bf16.h>

typedef __attribute__((ext_vector_type(16))) __bf16   v16bf;
typedef __attribute__((ext_vector_type(8)))  __bf16   v8bf;
typedef __attribute__((ext_vector_type(8)))  float    v8f;
typedef __attribute__((ext_vector_type(4)))  unsigned u32x4;
typedef __attribute__((ext_vector_type(8)))  int      i32x8;
typedef __attribute__((ext_vector_type(4)))  int      i32x4;

#define B_  4
#define T_  2048
#define C_  1024
#define H_  16
#define D_  64

static __device__ __forceinline__ v8f wmma_bf16(v16bf a, v16bf b, v8f c) {
  return __builtin_amdgcn_wmma_f32_16x16x32_bf16(false, a, false, b, (short)0, c, false, false);
}

// ---------------------------------------------------------------------------
// TDM descriptors (CDNA5 ISA), 6-arg builtin.
// Group 1 is loop-invariant (tile shape/stride/pad); group 0 carries the
// per-iteration LDS + global addresses. LDS row pad: 4 DW per 32 DW so a
// 64-bf16 row lands on a 72-element stride.
// ---------------------------------------------------------------------------
static __device__ __forceinline__ i32x8 tdm_g1_2d_bf16(
    unsigned tile_d0, unsigned tile_d1, unsigned long long stride_d0) {
  i32x8 g1;
  g1[0] = (int)((1u << 16) |            // data_size = 1 (2 bytes)
                (1u << 20) |            // pad_enable
                (4u << 22) |            // pad_interval: 32 DWORDs (128B row)
                (3u << 25));            // pad_amount: 4 DWORDs (16B)
  g1[1] = (int)((tile_d0 & 0xffffu) << 16);               // tensor_dim0
  g1[2] = (int)((tile_d1 & 0xffffu) << 16);               // tensor_dim1
  g1[3] = (int)((tile_d0 & 0xffffu) << 16);               // tile_dim0
  g1[4] = (int)(tile_d1 & 0xffffu);                       // tile_dim1
  g1[5] = (int)(unsigned)(stride_d0 & 0xffffffffu);       // dim0_stride lo
  g1[6] = (int)(unsigned)((stride_d0 >> 32) & 0xffffu);   // dim0_stride hi16
  g1[7] = 0;
  return g1;
}

static __device__ __forceinline__ void tdm_issue(
    unsigned lds_addr, const void* gaddr, i32x8 g1) {
  unsigned long long ga = (unsigned long long)(size_t)gaddr;
  u32x4 g0;
  g0[0] = 1u;                                                 // count=1
  g0[1] = lds_addr;
  g0[2] = (unsigned)(ga & 0xffffffffu);
  g0[3] = (unsigned)((ga >> 32) & 0x01ffffffu) | (2u << 30);  // type=2
  i32x4 z4 = {0, 0, 0, 0};
  i32x8 z8 = {0, 0, 0, 0, 0, 0, 0, 0};
  __builtin_amdgcn_tensor_load_to_lds(g0, g1, z4, z4, z8, 0);
}

// ---------------------------------------------------------------------------
// Kernel 1: qkv = x @ W_qkv + b_qkv (M=8192,N=3072,K=1024), fp32 in, bf16 out.
//   q -> [B][H][T][D] pre-scaled by 1/sqrt(D)
//   k -> [B][H][T][D]
//   v -> [B][H][D][T]   (transposed so the flash PV B-fragment is row-contiguous)
// ---------------------------------------------------------------------------
__global__ __launch_bounds__(256) void qkv_gemm_kernel(
    const float* __restrict__ X, const float* __restrict__ W,
    const float* __restrict__ bias, __bf16* __restrict__ qkv) {
  __shared__ __bf16 As[128][40];    // 128 x 32, row-major (k contiguous)
  __shared__ __bf16 BsT[128][40];   // 128(n) x 32(k), k contiguous

  const int tid  = threadIdx.x;
  const int lane = tid & 31;
  const int w    = tid >> 5;
  const int wm   = w >> 1;
  const int wn   = w & 1;
  const int half = lane >> 4;
  const int l16  = lane & 15;
  const int m0   = blockIdx.y * 128;
  const int n0   = blockIdx.x * 128;
  const int ncol = tid & 127;       // B-tile column owned by this thread
  const int kk   = (tid >> 7) * 16; // k half owned by this thread

  v8f acc[2][4];
#pragma unroll
  for (int mt = 0; mt < 2; ++mt)
#pragma unroll
    for (int nt = 0; nt < 4; ++nt) acc[mt][nt] = (v8f)(0.0f);

  for (int k0 = 0; k0 < C_; k0 += 32) {
    float4 ta[4];
#pragma unroll
    for (int r = 0; r < 4; ++r) {
      int idx = (tid + r * 256) * 4;
      int row = idx >> 5, col = idx & 31;
      ta[r] = *reinterpret_cast<const float4*>(X + (size_t)(m0 + row) * C_ + k0 + col);
    }
    float tb[16];
#pragma unroll
    for (int i = 0; i < 16; ++i)
      tb[i] = W[(size_t)(k0 + kk + i) * (3 * C_) + n0 + ncol];
    if (k0 + 32 < C_)
      __builtin_prefetch(W + (size_t)(k0 + 32 + kk) * (3 * C_) + n0 + ncol, 0, 1);

#pragma unroll
    for (int r = 0; r < 4; ++r) {
      int idx = (tid + r * 256) * 4;
      int row = idx >> 5, col = idx & 31;
      As[row][col + 0] = (__bf16)ta[r].x; As[row][col + 1] = (__bf16)ta[r].y;
      As[row][col + 2] = (__bf16)ta[r].z; As[row][col + 3] = (__bf16)ta[r].w;
    }
#pragma unroll
    for (int i = 0; i < 16; ++i) BsT[ncol][kk + i] = (__bf16)tb[i];
    __syncthreads();

    v16bf afrag[2], bfrag[4];
#pragma unroll
    for (int mt = 0; mt < 2; ++mt) {
      const __bf16* p = &As[wm * 32 + mt * 16 + l16][half * 16];
#pragma unroll
      for (int i = 0; i < 8; ++i) { afrag[mt][i] = p[i]; afrag[mt][8 + i] = p[8 + i]; }
    }
#pragma unroll
    for (int nt = 0; nt < 4; ++nt) {
      const __bf16* p = &BsT[wn * 64 + nt * 16 + l16][half * 16];
#pragma unroll
      for (int i = 0; i < 16; ++i) bfrag[nt][i] = p[i];
    }
#pragma unroll
    for (int mt = 0; mt < 2; ++mt)
#pragma unroll
      for (int nt = 0; nt < 4; ++nt)
        acc[mt][nt] = wmma_bf16(afrag[mt], bfrag[nt], acc[mt][nt]);
    __syncthreads();
  }

  const size_t sect = (size_t)B_ * H_ * T_ * D_;
#pragma unroll
  for (int mt = 0; mt < 2; ++mt)
#pragma unroll
    for (int nt = 0; nt < 4; ++nt) {
      const int n = n0 + wn * 64 + nt * 16 + l16;
      const int which = n >> 10, c = n & 1023, h = c >> 6, d = c & 63;
      const float bv = bias[n];
      const float sc = (which == 0) ? 0.125f : 1.0f;   // fold 1/sqrt(D) into q
#pragma unroll
      for (int r = 0; r < 8; ++r) {
        const int m = m0 + wm * 32 + mt * 16 + r + half * 8;
        const int b = m >> 11, t = m & 2047;
        const float val = (acc[mt][nt][r] + bv) * sc;
        size_t off;
        if (which == 2)      // v: [b][h][d][t]
          off = 2 * sect + (((size_t)(b * H_ + h) * D_ + d) * T_ + t);
        else                 // q/k: [b][h][t][d]
          off = (size_t)which * sect + (((size_t)(b * H_ + h) * T_ + t) * D_ + d);
        qkv[off] = (__bf16)val;
      }
    }
}

// ---------------------------------------------------------------------------
// Kernel 2: flash attention, double-buffered TDM pipeline.
// Block = 128 q-rows of one (b,h); 8 waves x 16 rows. While all waves compute
// on buffer jb&1, wave 0 has already launched the Tensor Data Mover on the
// jb+1 tiles into the other buffer; the s_wait_tensorcnt+barrier sits AFTER
// the compute so the DMA is fully hidden.
// ---------------------------------------------------------------------------
__global__ __launch_bounds__(256) void flash_attn_kernel(
    const __bf16* __restrict__ qkv, __bf16* __restrict__ Y) {
  __shared__ __bf16 Ks[2][64][72];     // [t][d]  (TDM pad -> 72 stride)
  __shared__ __bf16 Vs[2][64][72];     // [d][t]
  __shared__ __bf16 Ps[8][16][72];     // per-wave P staging

  const int tid  = threadIdx.x;
  const int lane = tid & 31;
  const int w    = tid >> 5;
  const int half = lane >> 4;
  const int l16  = lane & 15;
  const int bh   = blockIdx.y;
  const int bq   = blockIdx.x;
  const size_t sect  = (size_t)B_ * H_ * T_ * D_;
  const __bf16* Q = qkv + (size_t)bh * T_ * D_;
  const __bf16* K = qkv + sect + (size_t)bh * T_ * D_;
  const __bf16* V = qkv + 2 * sect + (size_t)bh * D_ * T_;   // [d][t]

  const int qbase = bq * 128 + w * 16;

  // Q fragments (16x64 = two 16x32 A-frags); already bf16 and pre-scaled
  v16bf qa[2];
#pragma unroll
  for (int kc = 0; kc < 2; ++kc) {
    const v8bf q0 = *reinterpret_cast<const v8bf*>(Q + (size_t)(qbase + l16) * D_ + kc * 32 + half * 16);
    const v8bf q1 = *reinterpret_cast<const v8bf*>(Q + (size_t)(qbase + l16) * D_ + kc * 32 + half * 16 + 8);
#pragma unroll
    for (int i = 0; i < 8; ++i) { qa[kc][i] = q0[i]; qa[kc][8 + i] = q1[i]; }
  }

  v8f o[4];
#pragma unroll
  for (int nt = 0; nt < 4; ++nt) o[nt] = (v8f)(0.0f);
  float mrow[8], lrow[8];
#pragma unroll
  for (int r = 0; r < 8; ++r) { mrow[r] = -3.0e38f; lrow[r] = 0.0f; }

  const i32x8 g1K = tdm_g1_2d_bf16(64, 64, (unsigned long long)D_);
  const i32x8 g1V = tdm_g1_2d_bf16(64, 64, (unsigned long long)T_);
  const unsigned ldsK0 = (unsigned)(size_t)&Ks[0][0][0];
  const unsigned ldsK1 = (unsigned)(size_t)&Ks[1][0][0];
  const unsigned ldsV0 = (unsigned)(size_t)&Vs[0][0][0];
  const unsigned ldsV1 = (unsigned)(size_t)&Vs[1][0][0];

  const int jmax = 2 * bq + 2;

  // prologue: fill buffer 0
  if (w == 0) {
    tdm_issue(ldsK0, K, g1K);
    tdm_issue(ldsV0, V, g1V);
  }
  __builtin_amdgcn_s_wait_tensorcnt((short)0);
  __syncthreads();

  for (int jb = 0; jb < jmax; ++jb) {
    const int cur = jb & 1;
    // launch DMA for next tiles into the other buffer (overlaps compute below)
    if (jb + 1 < jmax && w == 0) {
      tdm_issue(cur ? ldsK0 : ldsK1, K + (size_t)(jb + 1) * 64 * D_, g1K);
      tdm_issue(cur ? ldsV0 : ldsV1, V + (size_t)(jb + 1) * 64,      g1V);
    }

    // S = Q @ K^T : four 16x16 tiles (kcol chunks), contraction over d
    v8f s[4];
#pragma unroll
    for (int nt = 0; nt < 4; ++nt) {
      s[nt] = (v8f)(0.0f);
#pragma unroll
      for (int kc = 0; kc < 2; ++kc) {
        v16bf bf;
        const __bf16* kp = &Ks[cur][nt * 16 + l16][kc * 32 + half * 16];
#pragma unroll
        for (int i = 0; i < 16; ++i) bf[i] = kp[i];
        s[nt] = wmma_bf16(qa[kc], bf, s[nt]);
      }
    }

    if (jb * 64 + 63 > qbase) {          // causal mask (diagonal blocks only)
#pragma unroll
      for (int nt = 0; nt < 4; ++nt) {
        const int cg = jb * 64 + nt * 16 + l16;
#pragma unroll
        for (int r = 0; r < 8; ++r)
          if (cg > qbase + r + half * 8) s[nt][r] = -1.0e30f;
      }
    }

    // online softmax per row
#pragma unroll
    for (int r = 0; r < 8; ++r) {
      float mx = s[0][r];
#pragma unroll
      for (int nt = 1; nt < 4; ++nt) mx = fmaxf(mx, s[nt][r]);
#pragma unroll
      for (int off = 1; off < 16; off <<= 1) mx = fmaxf(mx, __shfl_xor(mx, off, 32));
      const float mnew  = fmaxf(mrow[r], mx);
      const float alpha = __expf(mrow[r] - mnew);
      float sum = 0.0f;
#pragma unroll
      for (int nt = 0; nt < 4; ++nt) {
        const float p = __expf(s[nt][r] - mnew);
        s[nt][r] = p; sum += p;
      }
#pragma unroll
      for (int off = 1; off < 16; off <<= 1) sum += __shfl_xor(sum, off, 32);
      lrow[r] = lrow[r] * alpha + sum;
      mrow[r] = mnew;
#pragma unroll
      for (int nt = 0; nt < 4; ++nt) o[nt][r] *= alpha;
      const int row = r + half * 8;
#pragma unroll
      for (int nt = 0; nt < 4; ++nt) Ps[w][row][nt * 16 + l16] = (__bf16)s[nt][r];
    }

    // O += P @ V : contraction over 64 kcols; Vs is [d][t] so reads are contiguous
#pragma unroll
    for (int nt = 0; nt < 4; ++nt) {
#pragma unroll
      for (int kc = 0; kc < 2; ++kc) {
        v16bf pa, vb;
        const __bf16* pp = &Ps[w][l16][kc * 32 + half * 16];
#pragma unroll
        for (int i = 0; i < 8; ++i) { pa[i] = pp[i]; pa[8 + i] = pp[8 + i]; }
        const __bf16* vp = &Vs[cur][nt * 16 + l16][kc * 32 + half * 16];
#pragma unroll
        for (int i = 0; i < 16; ++i) vb[i] = vp[i];
        o[nt] = wmma_bf16(pa, vb, o[nt]);
      }
    }

    // retire the in-flight DMA; all waves done reading `cur` before wave 0
    // overwrites it next iteration
    if (jb + 1 < jmax) {
      __builtin_amdgcn_s_wait_tensorcnt((short)0);
      __syncthreads();
    }
  }

  // epilogue: normalize, write y[b][h][t][d] (bf16)
#pragma unroll
  for (int r = 0; r < 8; ++r) {
    const float inv = 1.0f / lrow[r];
    const int t = qbase + r + half * 8;
#pragma unroll
    for (int nt = 0; nt < 4; ++nt)
      Y[(size_t)bh * T_ * D_ + (size_t)t * D_ + nt * 16 + l16] = (__bf16)(o[nt][r] * inv);
  }
}

// ---------------------------------------------------------------------------
// Kernel 3: out = y @ W_proj + b_proj (M=8192,N=1024,K=1024); y is bf16.
// ---------------------------------------------------------------------------
__global__ __launch_bounds__(256) void proj_gemm_kernel(
    const __bf16* __restrict__ Yw, const float* __restrict__ W,
    const float* __restrict__ bias, float* __restrict__ out) {
  __shared__ __bf16 As[128][40];
  __shared__ __bf16 BsT[128][40];

  const int tid  = threadIdx.x;
  const int lane = tid & 31;
  const int w    = tid >> 5;
  const int wm   = w >> 1;
  const int wn   = w & 1;
  const int half = lane >> 4;
  const int l16  = lane & 15;
  const int m0   = blockIdx.y * 128;
  const int n0   = blockIdx.x * 128;
  const int ncol = tid & 127;
  const int kk   = (tid >> 7) * 16;
  const int arow = tid >> 1;        // A-tile row owned by this thread
  const int ahk  = (tid & 1) * 16;  // A-tile k half

  v8f acc[2][4];
#pragma unroll
  for (int mt = 0; mt < 2; ++mt)
#pragma unroll
    for (int nt = 0; nt < 4; ++nt) acc[mt][nt] = (v8f)(0.0f);

  for (int k0 = 0; k0 < C_; k0 += 32) {
    const int m = m0 + arow, b = m >> 11, t = m & 2047;
    const int c = k0 + ahk, h = c >> 6, d = c & 63;
    const __bf16* yp = Yw + ((size_t)(b * H_ + h) * T_ + t) * D_ + d;
    v8bf a0 = *reinterpret_cast<const v8bf*>(yp);
    v8bf a1 = *reinterpret_cast<const v8bf*>(yp + 8);
    float tb[16];
#pragma unroll
    for (int i = 0; i < 16; ++i)
      tb[i] = W[(size_t)(k0 + kk + i) * C_ + n0 + ncol];
    if (k0 + 32 < C_)
      __builtin_prefetch(W + (size_t)(k0 + 32 + kk) * C_ + n0 + ncol, 0, 1);

    *reinterpret_cast<v8bf*>(&As[arow][ahk])     = a0;
    *reinterpret_cast<v8bf*>(&As[arow][ahk + 8]) = a1;
#pragma unroll
    for (int i = 0; i < 16; ++i) BsT[ncol][kk + i] = (__bf16)tb[i];
    __syncthreads();

    v16bf afrag[2], bfrag[4];
#pragma unroll
    for (int mt = 0; mt < 2; ++mt) {
      const __bf16* p = &As[wm * 32 + mt * 16 + l16][half * 16];
#pragma unroll
      for (int i = 0; i < 8; ++i) { afrag[mt][i] = p[i]; afrag[mt][8 + i] = p[8 + i]; }
    }
#pragma unroll
    for (int nt = 0; nt < 4; ++nt) {
      const __bf16* p = &BsT[wn * 64 + nt * 16 + l16][half * 16];
#pragma unroll
      for (int i = 0; i < 16; ++i) bfrag[nt][i] = p[i];
    }
#pragma unroll
    for (int mt = 0; mt < 2; ++mt)
#pragma unroll
      for (int nt = 0; nt < 4; ++nt)
        acc[mt][nt] = wmma_bf16(afrag[mt], bfrag[nt], acc[mt][nt]);
    __syncthreads();
  }

#pragma unroll
  for (int mt = 0; mt < 2; ++mt)
#pragma unroll
    for (int nt = 0; nt < 4; ++nt) {
      const int n = n0 + wn * 64 + nt * 16 + l16;
      const float bv = bias[n];
#pragma unroll
      for (int r = 0; r < 8; ++r) {
        const int m = m0 + wm * 32 + mt * 16 + r + half * 8;
        out[(size_t)m * C_ + n] = acc[mt][nt][r] + bv;
      }
    }
}

// ---------------------------------------------------------------------------
extern "C" void kernel_launch(void* const* d_in, const int* in_sizes, int n_in,
                              void* d_out, int out_size, void* d_ws, size_t ws_size,
                              hipStream_t stream) {
  const float* x      = (const float*)d_in[0];
  const float* W_qkv  = (const float*)d_in[1];
  const float* b_qkv  = (const float*)d_in[2];
  const float* W_proj = (const float*)d_in[3];
  const float* b_proj = (const float*)d_in[4];
  float* out = (float*)d_out;

  __bf16* qkv_ws = (__bf16*)d_ws;
  const size_t sect = (size_t)B_ * H_ * T_ * D_;
  __bf16* y_ws = qkv_ws + 3 * sect;

  qkv_gemm_kernel<<<dim3(3 * C_ / 128, B_ * T_ / 128), 256, 0, stream>>>(
      x, W_qkv, b_qkv, qkv_ws);
  flash_attn_kernel<<<dim3(T_ / 128, B_ * H_), 256, 0, stream>>>(qkv_ws, y_ws);
  proj_gemm_kernel<<<dim3(C_ / 128, B_ * T_ / 128), 256, 0, stream>>>(
      y_ws, W_proj, b_proj, out);
}
